// ConvFloodFillNetwork_609885356699
// MI455X (gfx1250) — matrix operand order repaired and checked by reading, not compiled, not run on hardware
//
#include <hip/hip_runtime.h>
#include <hip/hip_bf16.h>
#include <stdint.h>

// Problem constants (from reference)
#define B_  8
#define C_  256
#define F_  16384
#define NSTEPS 20
#define EPSN 1e-5f

typedef __attribute__((ext_vector_type(16))) __bf16 v16bf;
typedef __attribute__((ext_vector_type(8)))  float  v8f;

struct U8x32 { uint32_t d[8]; };

__device__ __forceinline__ unsigned short f2bf(float f) {
  uint32_t u = __float_as_uint(f);
  u += 0x7FFFu + ((u >> 16) & 1u);          // round-to-nearest-even
  return (unsigned short)(u >> 16);
}
__device__ __forceinline__ float bf2f(unsigned short s) {
  return __uint_as_float(((uint32_t)s) << 16);
}
__device__ __forceinline__ float sigmoidf_(float x) {
  return 1.0f / (1.0f + __expf(-x));
}
__device__ __forceinline__ v16bf mkfrag(uint4 a, uint4 b) {
  U8x32 u{{a.x, a.y, a.z, a.w, b.x, b.y, b.z, b.w}};
  return __builtin_bit_cast(v16bf, u);
}

// ---------------------------------------------------------------------------
// Pre-pack W_conv [1024,256] fp32 into hi/lo bf16 WMMA B-fragments.
// Layout: dword index = (((h*32 + kc)*16 + nt)*32 + lane)*8 + p
//   col = nt*16 + (lane&15); k = kc*32 + (lane>=16 ? 16 : 0) + 2*p (+1 in hi half)
// ---------------------------------------------------------------------------
__global__ __launch_bounds__(256) void pack_b_kernel(const float* __restrict__ W,
                                                     uint32_t* __restrict__ Bp) {
  int t = blockIdx.x * 256 + threadIdx.x;
  if (t >= 2 * 32 * 16 * 32 * 8) return;
  int p    = t & 7;
  int lane = (t >> 3) & 31;
  int nt   = (t >> 8) & 15;
  int kc   = (t >> 12) & 31;
  int h    = t >> 17;
  int col = nt * 16 + (lane & 15);
  int k   = kc * 32 + ((lane >> 4) * 16) + 2 * p;
  float w0 = W[(size_t)k * C_ + col];
  float w1 = W[(size_t)(k + 1) * C_ + col];
  unsigned short e0, e1;
  if (h == 0) {
    e0 = f2bf(w0); e1 = f2bf(w1);
  } else {
    e0 = f2bf(w0 - bf2f(f2bf(w0)));
    e1 = f2bf(w1 - bf2f(f2bf(w1)));
  }
  Bp[t] = (uint32_t)e0 | ((uint32_t)e1 << 16);
}

// ---------------------------------------------------------------------------
// GEMM (gathered A) + fused InstanceNorm + MLP(pred) epilogue.
// Block: 256 threads = 8 waves = 2 M-subtiles x 4 N-groups.
// Tile: 32 faces x 256 outputs, K = 1024 in 32-wide chunks (bf16x3 split).
// LDS A is double-buffered (stage kc+1 under the WMMAs of kc, one barrier
// per chunk); staging packs hi/lo into ds_store_b64 (2 DS ops/thread/chunk).
// ---------------------------------------------------------------------------
__global__ __launch_bounds__(256) void gemm_kernel(
    const float* __restrict__ x, const int* __restrict__ adj,
    const uint32_t* __restrict__ Bp, const float* __restrict__ b_conv,
    const float* __restrict__ W_mlp, const float* __restrict__ b_mlp,
    float* __restrict__ outF, float* __restrict__ pred) {
  __shared__ __align__(16) unsigned short sAh[2][32 * 40];  // row stride 40 shorts (80B, 16B-aligned)
  __shared__ __align__(16) unsigned short sAl[2][32 * 40];
  __shared__ float redS[4][32];
  __shared__ float redQ[4][32];
  __shared__ float redD[4][32];

  const int t      = threadIdx.x;
  const int lane   = t & 31;
  const int wave   = t >> 5;
  const int wave_m = wave >> 2;     // 0..1
  const int wave_n = wave & 3;      // 0..3
  const int laneHalf = lane >> 4;   // 0/1
  const int ln     = lane & 15;

  const int tile = blockIdx.x;
  const int bidx = tile >> 9;            // F/32 = 512 tiles per mesh
  const int f0   = (tile & 511) << 5;
  const int fi   = t & 31;               // staging face-in-tile
  const int p4   = (t >> 5) << 2;        // staging channel group: 4 consecutive K

  const float* xb = x + (size_t)bidx * C_ * F_;

  // Hoisted gather indices: one source face per K-segment (self + 3 neighbors)
  const int* arow = adj + ((size_t)(bidx * F_ + f0 + fi)) * 3;
  const int gs0 = f0 + fi;
  const int gs1 = arow[0];
  const int gs2 = arow[1];
  const int gs3 = arow[2];

  auto stage = [&](int kc, unsigned short* Ah, unsigned short* Al) {
    int seg = kc >> 3;                   // 0=self, 1..3=neighbors
    int g = (seg == 0) ? gs0 : (seg == 1) ? gs1 : (seg == 2) ? gs2 : gs3;
    unsigned short hi[4], lo[4];
#pragma unroll
    for (int j = 0; j < 4; j++) {
      int c = ((kc & 7) << 5) + p4 + j;          // channel within segment
      float v = xb[(size_t)c * F_ + g];
      hi[j] = f2bf(v);
      lo[j] = f2bf(v - bf2f(hi[j]));
    }
    uint2 hw, lw;
    hw.x = (uint32_t)hi[0] | ((uint32_t)hi[1] << 16);
    hw.y = (uint32_t)hi[2] | ((uint32_t)hi[3] << 16);
    lw.x = (uint32_t)lo[0] | ((uint32_t)lo[1] << 16);
    lw.y = (uint32_t)lo[2] | ((uint32_t)lo[3] << 16);
    *(uint2*)&Ah[fi * 40 + p4] = hw;             // byte addr fi*80 + p4*2, 8B-aligned
    *(uint2*)&Al[fi * 40 + p4] = lw;
  };

  v8f acc[4] = {};
  const int k0 = laneHalf ? 16 : 0;      // contiguous K origin per half-wave (matches packed B)
  const int rowoff = (wave_m * 16 + ln) * 40 + k0;

  stage(0, sAh[0], sAl[0]);
  __syncthreads();

  for (int kc = 0; kc < 32; kc++) {
    int cur = kc & 1;
    if (kc + 1 < 32) stage(kc + 1, sAh[cur ^ 1], sAl[cur ^ 1]);

    // ---- A fragments: contiguous K (k = 16*half + s), same order as packed B ----
    const unsigned short* ra = &sAh[cur][rowoff];
    const unsigned short* rb = &sAl[cur][rowoff];
    v16bf ah = mkfrag(*(const uint4*)ra, *(const uint4*)(ra + 8));
    v16bf al = mkfrag(*(const uint4*)rb, *(const uint4*)(rb + 8));

#pragma unroll
    for (int nt = 0; nt < 4; nt++) {
      size_t offh = ((((size_t)0 * 32 + kc) * 16 + (wave_n * 4 + nt)) * 32 + lane) * 8;
      size_t offl = ((((size_t)1 * 32 + kc) * 16 + (wave_n * 4 + nt)) * 32 + lane) * 8;
      v16bf bh = mkfrag(*(const uint4*)(Bp + offh), *(const uint4*)(Bp + offh + 4));
      v16bf bl = mkfrag(*(const uint4*)(Bp + offl), *(const uint4*)(Bp + offl + 4));
      // bf16x3: hi*hi + lo*hi + hi*lo, fp32 accumulate
      acc[nt] = __builtin_amdgcn_wmma_f32_16x16x32_bf16(false, ah, false, bh, (short)0, acc[nt], false, false);
      acc[nt] = __builtin_amdgcn_wmma_f32_16x16x32_bf16(false, al, false, bh, (short)0, acc[nt], false, false);
      acc[nt] = __builtin_amdgcn_wmma_f32_16x16x32_bf16(false, ah, false, bl, (short)0, acc[nt], false, false);
    }
    if (kc + 1 < 32)
      __builtin_prefetch(Bp + (((size_t)(kc + 1) * 16 + wave_n * 4) * 32 + lane) * 8, 0, 0);
    __syncthreads();
  }

  // ---- epilogue: + b_conv, InstanceNorm over C=256, pred = sigmoid(h@W_mlp+b) ----
  float hv[4][8];
#pragma unroll
  for (int nt = 0; nt < 4; nt++) {
    float bc = b_conv[wave_n * 64 + nt * 16 + ln];
#pragma unroll
    for (int r = 0; r < 8; r++) hv[nt][r] = acc[nt][r] + bc;
  }

#pragma unroll
  for (int r = 0; r < 8; r++) {
    float s = 0.f, q = 0.f;
#pragma unroll
    for (int nt = 0; nt < 4; nt++) { float v = hv[nt][r]; s += v; q += v * v; }
#pragma unroll
    for (int m = 1; m < 16; m <<= 1) { s += __shfl_xor(s, m, 32); q += __shfl_xor(q, m, 32); }
    if (ln == 0) {
      int face = wave_m * 16 + (laneHalf ? 8 + r : r);
      redS[wave_n][face] = s;
      redQ[wave_n][face] = q;
    }
  }
  __syncthreads();

  float wm[4];
#pragma unroll
  for (int nt = 0; nt < 4; nt++) wm[nt] = W_mlp[wave_n * 64 + nt * 16 + ln];

#pragma unroll
  for (int r = 0; r < 8; r++) {
    int face = wave_m * 16 + (laneHalf ? 8 + r : r);
    float s = redS[0][face] + redS[1][face] + redS[2][face] + redS[3][face];
    float q = redQ[0][face] + redQ[1][face] + redQ[2][face] + redQ[3][face];
    float mean = s * (1.0f / 256.0f);
    float var  = q * (1.0f / 256.0f) - mean * mean;
    float inv  = rsqrtf(var + EPSN);
    float d = 0.f;
#pragma unroll
    for (int nt = 0; nt < 4; nt++) {
      float v = (hv[nt][r] - mean) * inv;
      hv[nt][r] = v;
      d += v * wm[nt];
    }
#pragma unroll
    for (int m = 1; m < 16; m <<= 1) d += __shfl_xor(d, m, 32);
    if (ln == 0) redD[wave_n][face] = d;
  }
  __syncthreads();

  if (wave_n == 0 && ln == 0) {
    float bm = b_mlp[0];
#pragma unroll
    for (int r = 0; r < 8; r++) {
      int face = wave_m * 16 + (laneHalf ? 8 + r : r);
      float tot = redD[0][face] + redD[1][face] + redD[2][face] + redD[3][face] + bm;
      pred[(size_t)bidx * F_ + f0 + face] = sigmoidf_(tot);
    }
  }

  // store normalized h into out features (done-faces keep it; others fixed up later)
#pragma unroll
  for (int r = 0; r < 8; r++) {
    int face = wave_m * 16 + (laneHalf ? 8 + r : r);
    float* op = outF + ((size_t)(bidx * F_ + f0 + face)) * C_ + wave_n * 64 + ln;
#pragma unroll
    for (int nt = 0; nt < 4; nt++) op[nt * 16] = hv[nt][r];
  }
}

// ---------------------------------------------------------------------------
// init_scores = sigmoid(feat @ W_mlp + b); seed BFS state at anchors.
// ---------------------------------------------------------------------------
__global__ __launch_bounds__(256) void init_kernel(
    const float* __restrict__ x, const float* __restrict__ W_mlp,
    const float* __restrict__ b_mlp, const int* __restrict__ anchors,
    float* __restrict__ scores, int* __restrict__ done,
    int* __restrict__ boundary, int* __restrict__ reach) {
  int tid = blockIdx.x * 256 + threadIdx.x;
  if (tid >= B_ * F_) return;
  int b = tid >> 14;
  int f = tid & (F_ - 1);
  const float* xb = x + (size_t)b * C_ * F_ + f;
  float s = 0.f;
#pragma unroll 8
  for (int c = 0; c < C_; c++) s += xb[(size_t)c * F_] * W_mlp[c];
  scores[tid] = sigmoidf_(s + b_mlp[0]);
  int d = (f == anchors[b]) ? 1 : 0;
  done[tid] = d;
  boundary[tid] = d;
  reach[tid] = 0;
}

// ---------------------------------------------------------------------------
// BFS step, phase A: boundary faces update scores (reads only frozen "done"
// neighbors -> in-place safe) and scatter the next frontier.
// ---------------------------------------------------------------------------
__global__ __launch_bounds__(256) void stepA_kernel(
    const int* __restrict__ adj, const float* __restrict__ pred,
    float* __restrict__ scores, const int* __restrict__ done,
    const int* __restrict__ boundary, int* __restrict__ reach) {
  int tid = blockIdx.x * 256 + threadIdx.x;
  if (tid >= B_ * F_) return;
  if (!boundary[tid]) return;
  int base = (tid >> 14) << 14;
  const int* ap = adj + (size_t)tid * 3;
  int a0 = ap[0], a1 = ap[1], a2 = ap[2];
  float mx = -1.0f;
  bool any = false;
#pragma unroll
  for (int k = 0; k < 3; k++) {
    int gi = base + ((k == 0) ? a0 : (k == 1) ? a1 : a2);
    if (done[gi]) { any = true; mx = fmaxf(mx, scores[gi]); }
  }
  float nbmax = any ? mx : 1.0f;
  float nw = fminf(fmaxf(pred[tid], scores[tid]), nbmax);
  scores[tid] = nw;
  atomicOr(&reach[base + a0], 1);
  atomicOr(&reach[base + a1], 1);
  atomicOr(&reach[base + a2], 1);
}

// Phase B: fold boundary into done, compute next frontier, clear reach.
__global__ __launch_bounds__(256) void stepB_kernel(
    int* __restrict__ done, int* __restrict__ boundary, int* __restrict__ reach) {
  int tid = blockIdx.x * 256 + threadIdx.x;
  if (tid >= B_ * F_) return;
  int d = done[tid] | boundary[tid];
  int nb = (reach[tid] && !d) ? 1 : 0;
  done[tid] = d;
  boundary[tid] = nb;
  reach[tid] = 0;
}

// ---------------------------------------------------------------------------
// Finalize: scores out; un-visited faces get original (transposed) features.
// ---------------------------------------------------------------------------
__global__ __launch_bounds__(256) void final_kernel(
    const float* __restrict__ x, const float* __restrict__ scores,
    const int* __restrict__ done, float* __restrict__ outF,
    float* __restrict__ outS) {
  int tid = blockIdx.x * 256 + threadIdx.x;
  if (tid >= B_ * F_) return;
  outS[tid] = scores[tid];
  if (!done[tid]) {
    int b = tid >> 14;
    int f = tid & (F_ - 1);
    const float* xp = x + (size_t)b * C_ * F_ + f;
    float* op = outF + (size_t)tid * C_;
#pragma unroll 4
    for (int c = 0; c < C_; c++) op[c] = xp[(size_t)c * F_];
  }
}

// ---------------------------------------------------------------------------
extern "C" void kernel_launch(void* const* d_in, const int* in_sizes, int n_in,
                              void* d_out, int out_size, void* d_ws, size_t ws_size,
                              hipStream_t stream) {
  (void)in_sizes; (void)n_in; (void)out_size; (void)ws_size;
  const float* x       = (const float*)d_in[0];
  const int*   adj     = (const int*)d_in[1];
  const int*   anchors = (const int*)d_in[2];
  const float* W_conv  = (const float*)d_in[3];
  const float* b_conv  = (const float*)d_in[4];
  const float* W_mlp   = (const float*)d_in[5];
  const float* b_mlp   = (const float*)d_in[6];

  float* outF = (float*)d_out;
  float* outS = outF + (size_t)B_ * F_ * C_;

  char* ws = (char*)d_ws;
  uint32_t* Bp   = (uint32_t*)ws;                                   // 1 MB packed B frags
  float* pred    = (float*)(ws + (1u << 20));                       // 512 KB
  float* scores  = (float*)(ws + (1u << 20) + 1 * (512u << 10));    // 512 KB
  int*   done    = (int*)  (ws + (1u << 20) + 2 * (512u << 10));
  int*   boundary= (int*)  (ws + (1u << 20) + 3 * (512u << 10));
  int*   reach   = (int*)  (ws + (1u << 20) + 4 * (512u << 10));

  pack_b_kernel<<<1024, 256, 0, stream>>>(W_conv, Bp);
  gemm_kernel<<<B_ * (F_ / 32), 256, 0, stream>>>(x, adj, Bp, b_conv, W_mlp, b_mlp, outF, pred);
  init_kernel<<<(B_ * F_) / 256, 256, 0, stream>>>(x, W_mlp, b_mlp, anchors, scores, done, boundary, reach);
  for (int i = 0; i < NSTEPS; i++) {
    stepA_kernel<<<(B_ * F_) / 256, 256, 0, stream>>>(adj, pred, scores, done, boundary, reach);
    stepB_kernel<<<(B_ * F_) / 256, 256, 0, stream>>>(done, boundary, reach);
  }
  final_kernel<<<(B_ * F_) / 256, 256, 0, stream>>>(x, scores, done, outF, outS);
}
